// KronEmbed_70686571757754
// MI455X (gfx1250) — matrix operand order repaired and checked by reading, not compiled, not run on hardware
//
#include <hip/hip_runtime.h>
#include <hip/hip_bf16.h>

typedef __attribute__((ext_vector_type(16))) _Float16 v16h;
typedef __attribute__((ext_vector_type(8)))  _Float16 h8;
typedef __attribute__((ext_vector_type(8)))  float    v8f;

#define NB   32
#define CCH  256
#define HWP  1024   // H*W

// ---------------------------------------------------------------------------
// Kernel 1: inverse L2 norms over channels for both tensors.
// idx -> {tensor t, batch n, position pos}; consecutive threads -> consecutive
// pos so every c-iteration is a fully coalesced 128B line per wave.
// ---------------------------------------------------------------------------
__global__ __launch_bounds__(256) void norm_kernel(
    const float* __restrict__ x1, const float* __restrict__ x2,
    float* __restrict__ inv1, float* __restrict__ inv2) {
  int idx = blockIdx.x * 256 + threadIdx.x;        // 0 .. 65535
  int t   = idx >> 15;
  int rem = idx & 32767;
  int n   = rem >> 10;
  int pos = rem & 1023;
  const float* x = t ? x2 : x1;
  const float* p = x + (size_t)n * CCH * HWP + pos;
  float acc = 0.f;
#pragma unroll 8
  for (int c = 0; c < CCH; ++c) {
    float v = p[(size_t)c * HWP];
    acc += v * v;
  }
  float* dst = t ? inv2 : inv1;
  dst[n * HWP + pos] = rsqrtf(acc);
}

// ---------------------------------------------------------------------------
// Kernel 2: build f16 operands.
//   Qh[n,p,c] = x1[n,c,p] * inv1[n,p]   (transpose via LDS tile)
//   Kh[n,q,c] = x2[n,c,q] * inv2[n,q]   (transpose via LDS tile)
//   Vh[n,c,q] = x2[n,c,q]               (straight convert)
// ---------------------------------------------------------------------------
__global__ __launch_bounds__(256) void pack_kernel(
    const float* __restrict__ x1, const float* __restrict__ x2,
    const float* __restrict__ inv1, const float* __restrict__ inv2,
    _Float16* __restrict__ Qh, _Float16* __restrict__ Kh,
    _Float16* __restrict__ Vh) {
  __shared__ float tile[32][33];
  const int posb = blockIdx.x * 32;
  const int cb   = blockIdx.y * 32;
  const int nz   = blockIdx.z;            // n*2 + t
  const int n    = nz >> 1;
  const int t    = nz & 1;
  const float*   x   = t ? x2 : x1;
  const float*   inv = t ? inv2 : inv1;
  _Float16*      dst = t ? Kh : Qh;
  const int tx = threadIdx.x;             // 0..31
  const int ty = threadIdx.y;             // 0..7
  const size_t base = (size_t)n * CCH * HWP;
#pragma unroll
  for (int r = 0; r < 4; ++r) {
    int c = cb + ty + r * 8;
    float v = x[base + (size_t)c * HWP + posb + tx];
    tile[ty + r * 8][tx] = v;
    if (t) Vh[base + (size_t)c * HWP + posb + tx] = (_Float16)v;
  }
  __syncthreads();
#pragma unroll
  for (int r = 0; r < 4; ++r) {
    int pos = posb + ty + r * 8;
    float sc = inv[n * HWP + pos];
    dst[((size_t)n * HWP + pos) * CCH + cb + tx] =
        (_Float16)(tile[tx][ty + r * 8] * sc);
  }
}

// ---------------------------------------------------------------------------
// Kernel 3: fused attention.  One workgroup = (batch n, 32 p-rows).
// 8 waves: pg = wave>>2 selects p-group of 16 rows, j = wave&3 selects the
// q-tile it scores and the 64-channel output slice it accumulates.
// Scores are in [-1,1] (unit vectors) so softmax needs no max subtraction:
// accumulate exp(S)@V and row sums, divide once at the end.
// ---------------------------------------------------------------------------
__global__ __launch_bounds__(256) void attn_kernel(
    const _Float16* __restrict__ Qh, const _Float16* __restrict__ Kh,
    const _Float16* __restrict__ Vh, float* __restrict__ out) {
  __shared__ _Float16 Pbuf[2][16][72];   // padded rows: conflict-free b128 reads
  __shared__ float    rsLDS[2][16];
  __shared__ float    OutS[256][33];     // [c][p] staging for coalesced stores

  const int tid  = threadIdx.x;
  const int lane = tid & 31;
  const int w    = tid >> 5;
  const int pg   = w >> 2;               // 0..1
  const int j    = w & 3;                // 0..3
  const int n    = blockIdx.y;
  const int pwg  = blockIdx.x * 32;
  const int lhi  = lane >> 4;            // 0 or 1
  const int llo  = lane & 15;

  // Preload full A (16 p-rows x 256 channels) for this p-group: 8 fragments.
  v16h Afrag[8];
  {
    const _Float16* qrow =
        Qh + ((size_t)(n * HWP + pwg + pg * 16 + llo)) * CCH;
#pragma unroll
    for (int ck = 0; ck < 8; ++ck) {
      const _Float16* p0 = qrow + ck * 32 + lhi * 16;  // K = k0 .. k0+7
      h8 lo = *reinterpret_cast<const h8*>(p0);
      h8 hi = *reinterpret_cast<const h8*>(p0 + 8);    // K = k0+8 .. k0+15
      Afrag[ck] = __builtin_shufflevector(lo, hi, 0, 1, 2, 3, 4, 5, 6, 7,
                                          8, 9, 10, 11, 12, 13, 14, 15);
    }
  }

  v8f Oacc[4] = {{}, {}, {}, {}};
  float rs = 0.f;

  for (int it = 0; it < 16; ++it) {
    const int qb = it * 64;

    // ---- S = Qh * Kh^T for q-tile (qb + j*16): 8 WMMAs over K=256
    v8f s = {};
    const _Float16* krow =
        Kh + ((size_t)(n * HWP + qb + j * 16 + llo)) * CCH + lhi * 16;
#pragma unroll
    for (int ck = 0; ck < 8; ++ck) {
      v16h b = *reinterpret_cast<const v16h*>(krow + ck * 32);
      s = __builtin_amdgcn_wmma_f32_16x16x32_f16(false, Afrag[ck], false, b,
                                                 (short)0, s, false, false);
    }

    __syncthreads();  // previous iteration's Pbuf consumers are done
#pragma unroll
    for (int r = 0; r < 8; ++r) {
      // D layout: row = r + 8*lhi, col = llo
      Pbuf[pg][r + lhi * 8][j * 16 + llo] = (_Float16)__expf(s[r]);
    }
    __syncthreads();  // Pbuf ready for all 4 waves of the p-group

    // ---- running softmax denominator (one wave per p-group)
    if (j == 0) {
      float acc = 0.f;
#pragma unroll
      for (int q = 0; q < 32; ++q)
        acc += (float)Pbuf[pg][llo][lhi * 32 + q];
      rs += acc;
    }

    // ---- O += P * V  (2 K-chunks of 32 q  x  4 c-tiles)
#pragma unroll
    for (int kc = 0; kc < 2; ++kc) {
      const _Float16* prow = &Pbuf[pg][llo][kc * 32 + lhi * 16];
      h8 lo = *reinterpret_cast<const h8*>(prow);
      h8 hi = *reinterpret_cast<const h8*>(prow + 8);
      v16h a2 = __builtin_shufflevector(lo, hi, 0, 1, 2, 3, 4, 5, 6, 7,
                                        8, 9, 10, 11, 12, 13, 14, 15);
#pragma unroll
      for (int ct = 0; ct < 4; ++ct) {
        const _Float16* vrow =
            Vh + ((size_t)(n * CCH + j * 64 + ct * 16 + llo)) * HWP +
            qb + kc * 32 + lhi * 16;
        v16h b = *reinterpret_cast<const v16h*>(vrow);
        Oacc[ct] = __builtin_amdgcn_wmma_f32_16x16x32_f16(
            false, a2, false, b, (short)0, Oacc[ct], false, false);
      }
    }
  }

  // ---- finalize row sums
  if (j == 0) {
    rs += __shfl_xor(rs, 16, 32);
    if (lane < 16) rsLDS[pg][lane] = rs;
  }
  __syncthreads();

  // ---- divide and stage transposed [c][p]
#pragma unroll
  for (int r = 0; r < 8; ++r) {
    const int prow = pg * 16 + r + lhi * 8;
    const float inv = 1.f / rsLDS[pg][r + lhi * 8];
#pragma unroll
    for (int ct = 0; ct < 4; ++ct)
      OutS[j * 64 + ct * 16 + llo][prow] = Oacc[ct][r] * inv;
  }
  __syncthreads();

  // ---- coalesced store: 32 consecutive p per row of c
  float* obase = out + (size_t)n * CCH * HWP + pwg;
#pragma unroll 4
  for (int k = tid; k < 256 * 32; k += 256) {
    obase[(size_t)(k >> 5) * HWP + (k & 31)] = OutS[k >> 5][k & 31];
  }
}

// ---------------------------------------------------------------------------
extern "C" void kernel_launch(void* const* d_in, const int* in_sizes, int n_in,
                              void* d_out, int out_size, void* d_ws, size_t ws_size,
                              hipStream_t stream) {
  const float* x1 = (const float*)d_in[0];
  const float* x2 = (const float*)d_in[1];
  float* out = (float*)d_out;

  // workspace: inv1 | inv2 | Qh | Kh | Vh   (~48.25 MiB, all 32B aligned)
  float* inv1 = (float*)d_ws;
  float* inv2 = inv1 + NB * HWP;
  _Float16* Qh = (_Float16*)(inv2 + NB * HWP);
  _Float16* Kh = Qh + (size_t)NB * HWP * CCH;
  _Float16* Vh = Kh + (size_t)NB * HWP * CCH;

  norm_kernel<<<256, 256, 0, stream>>>(x1, x2, inv1, inv2);
  pack_kernel<<<dim3(32, 8, 64), dim3(32, 8), 0, stream>>>(x1, x2, inv1, inv2,
                                                           Qh, Kh, Vh);
  attn_kernel<<<dim3(32, 32), 256, 0, stream>>>(Qh, Kh, Vh, out);
}
